// MultiheadAttention_10299331576131
// MI455X (gfx1250) — matrix operand order repaired and checked by reading, not compiled, not run on hardware
//
#include <hip/hip_runtime.h>
#include <hip/hip_bf16.h>

// MI455X / gfx1250, wave32. bf16 WMMA (f32 accum) end-to-end MHA pipeline.
// Round 4: hardware packed f32->bf16 conversion (v_cvt_pk_bf16_f32) to cut the
// staging VALU overhead; TDM K-block loads + async LDS copies unchanged.

typedef __attribute__((ext_vector_type(16))) __bf16 v16bf;
typedef __attribute__((ext_vector_type(8)))  __bf16 v8bf;
typedef __attribute__((ext_vector_type(8)))  float  v8f;
typedef __attribute__((ext_vector_type(4)))  unsigned int v4u;
typedef __attribute__((ext_vector_type(8)))  int v8i;
typedef __attribute__((ext_vector_type(4)))  int v4i;

#define DEVINL __device__ __forceinline__

constexpr int Bc = 4, Sc = 2048, Ec = 1024, Hc = 16, Dc = 64;
constexpr int Mtot = Bc * Sc;   // 8192
constexpr int KDIM = Ec;        // reduction dim of both GEMMs

DEVINL unsigned short f2bf(float f) {
  unsigned u = __float_as_uint(f);
  unsigned r = u + 0x7FFFu + ((u >> 16) & 1u);   // round-to-nearest-even
  return (unsigned short)(r >> 16);
}

#if __has_builtin(__builtin_amdgcn_cvt_pk_bf16_f32)
DEVINL unsigned pk2(float a, float b) {
  auto r = __builtin_amdgcn_cvt_pk_bf16_f32(a, b);   // v_cvt_pk_bf16_f32 (RNE)
  unsigned u;
  __builtin_memcpy(&u, &r, 4);
  return u;
}
#else
DEVINL unsigned pk2(float a, float b) {
  return (unsigned)f2bf(a) | ((unsigned)f2bf(b) << 16);
}
#endif

DEVINL v8f vzero8() {
  v8f z;
#pragma unroll
  for (int i = 0; i < 8; i++) z[i] = 0.0f;
  return z;
}
// Assemble a 16x32 A-fragment lane payload from two contiguous 8-element chunks
// (as used here, contiguous K: lanes 0-15 hold K 0..15, lanes 16-31 K 16..31).
DEVINL v16bf cat8(v8bf lo, v8bf hi) {
  v16bf f;
#pragma unroll
  for (int e = 0; e < 8; e++) { f[e] = lo[e]; f[e + 8] = hi[e]; }
  return f;
}
DEVINL v8f bfwmma(v16bf a, v16bf b, v8f c) {
  return __builtin_amdgcn_wmma_f32_16x16x32_bf16(false, a, false, b, (short)0, c,
                                                 false, false);
}
// Low 32 bits of a flat shared-memory address are the LDS byte offset
// (LDS aperture occupies addr[63:32]).
DEVINL unsigned lds_off(const void* p) {
  return (unsigned)(uintptr_t)p;
}

// ---------------------------------------------------------------------------
// Generic 128x128-tile GEMM body: C = A(MxK) * W(NxK)^T.
// A is f32 (converted while staging) or bf16 (async-DMA'd to LDS); W is f32.
// OUT_HEAD: store bf16 into head-major [B,H,S,D]; else f32 row-major [M,N].
// 256 threads = 8 waves; wave (w&3, w>>2) owns a 32x64 sub-tile (2x4 WMMA frags).
// ---------------------------------------------------------------------------
template <bool A_BF16, bool OUT_HEAD>
DEVINL void gemm_body(const void* __restrict__ Ap, const float* __restrict__ Wp,
                      void* __restrict__ Cp) {
  __shared__ unsigned short As[128 * 32];
  __shared__ unsigned short Bs[128 * 32];

  const int t = threadIdx.x, lane = t & 31, w = t >> 5;
  const int lm = lane & 15, hi = lane >> 4;
  const int wm = (w & 3) * 32, wn = (w >> 2) * 64;
  const int mTile = blockIdx.x * 128, nTile = blockIdx.y * 128;
  const int r = t >> 1, cb = (t & 1) * 16;   // staging: each thread = 16 elems of a row

  v8f acc[2][4];
#pragma unroll
  for (int i = 0; i < 2; i++)
#pragma unroll
    for (int j = 0; j < 4; j++) acc[i][j] = vzero8();

  for (int k0 = 0; k0 < KDIM; k0 += 32) {
    // ---- stage A tile (128x32) into bf16 LDS ----
    if (A_BF16) {
      // pure byte copy: route it through the async global->LDS path (ASYNCcnt)
      const unsigned short* a =
          (const unsigned short*)Ap + (size_t)(mTile + r) * KDIM + k0 + cb;
      const unsigned l0 = lds_off(&As[r * 32 + cb]);
      asm volatile(
          "global_load_async_to_lds_b128 %0, %2, off\n\t"
          "global_load_async_to_lds_b128 %1, %3, off"
          :
          : "v"(l0), "v"(l0 + 16u), "v"(a), "v"(a + 8)
          : "memory");
      if (k0 + 32 < KDIM) __builtin_prefetch(a + 32, 0, 1);
    } else {
      const float* a = (const float*)Ap + (size_t)(mTile + r) * KDIM + k0 + cb;
      float4 f0 = *(const float4*)(a + 0);
      float4 f1 = *(const float4*)(a + 4);
      float4 f2 = *(const float4*)(a + 8);
      float4 f3 = *(const float4*)(a + 12);
      uint4 p0 = {pk2(f0.x, f0.y), pk2(f0.z, f0.w), pk2(f1.x, f1.y), pk2(f1.z, f1.w)};
      uint4 p1 = {pk2(f2.x, f2.y), pk2(f2.z, f2.w), pk2(f3.x, f3.y), pk2(f3.z, f3.w)};
      *(uint4*)&As[r * 32 + cb]     = p0;
      *(uint4*)&As[r * 32 + cb + 8] = p1;
      if (k0 + 32 < KDIM) __builtin_prefetch(a + 32, 0, 1);
    }
    // ---- stage W tile (128 output-cols x 32 k) into bf16 LDS ----
    {
      const float* b = Wp + (size_t)(nTile + r) * KDIM + k0 + cb;
      float4 f0 = *(const float4*)(b + 0);
      float4 f1 = *(const float4*)(b + 4);
      float4 f2 = *(const float4*)(b + 8);
      float4 f3 = *(const float4*)(b + 12);
      uint4 p0 = {pk2(f0.x, f0.y), pk2(f0.z, f0.w), pk2(f1.x, f1.y), pk2(f1.z, f1.w)};
      uint4 p1 = {pk2(f2.x, f2.y), pk2(f2.z, f2.w), pk2(f3.x, f3.y), pk2(f3.z, f3.w)};
      *(uint4*)&Bs[r * 32 + cb]     = p0;
      *(uint4*)&Bs[r * 32 + cb + 8] = p1;
      if (k0 + 32 < KDIM) __builtin_prefetch(b + 32, 0, 1);
    }
    if (A_BF16) asm volatile("s_wait_asynccnt 0x0" ::: "memory");
    __syncthreads();

    // ---- fragments + WMMA ----
    v16bf af[2];
#pragma unroll
    for (int i = 0; i < 2; i++) {
      const unsigned short* ap = &As[(wm + i * 16 + lm) * 32 + hi * 16];
      af[i] = cat8(*(const v8bf*)ap, *(const v8bf*)(ap + 8));
    }
#pragma unroll
    for (int j = 0; j < 4; j++) {
      // B frag (32x16): lane = column N, 16 contiguous K at (lane/16)*16
      v16bf bf = *(const v16bf*)&Bs[(wn + j * 16 + lm) * 32 + hi * 16];
#pragma unroll
      for (int i = 0; i < 2; i++) acc[i][j] = bfwmma(af[i], bf, acc[i][j]);
    }
    __syncthreads();
  }

  // ---- store (C layout: VGPR e -> M = e + 8*hi, N = lm) ----
#pragma unroll
  for (int i = 0; i < 2; i++)
#pragma unroll
    for (int j = 0; j < 4; j++)
#pragma unroll
      for (int e = 0; e < 8; e++) {
        const int m = mTile + wm + i * 16 + e + hi * 8;
        const int n = nTile + wn + j * 16 + lm;
        const float val = acc[i][j][e];
        if (OUT_HEAD) {
          const int bb = m >> 11, ss = m & (Sc - 1);
          const int hh = n >> 6, dd = n & (Dc - 1);
          ((unsigned short*)Cp)[(((size_t)(bb * Hc + hh)) * Sc + ss) * Dc + dd] =
              f2bf(val);
        } else {
          ((float*)Cp)[(size_t)m * Ec + n] = val;
        }
      }
}

__global__ __launch_bounds__(256) void proj_qkv_kernel(
    const float* __restrict__ q, const float* __restrict__ k,
    const float* __restrict__ v, const float* __restrict__ Wq,
    const float* __restrict__ Wk, const float* __restrict__ Wv,
    unsigned short* __restrict__ Qo, unsigned short* __restrict__ Ko,
    unsigned short* __restrict__ Vo) {
  const float* A; const float* W; unsigned short* C;
  if (blockIdx.z == 0)      { A = q; W = Wq; C = Qo; }
  else if (blockIdx.z == 1) { A = k; W = Wk; C = Ko; }
  else                      { A = v; W = Wv; C = Vo; }
  gemm_body<false, true>(A, W, C);
}

__global__ __launch_bounds__(256) void out_proj_kernel(
    const unsigned short* __restrict__ ctx, const float* __restrict__ Wo,
    float* __restrict__ out) {
  gemm_body<true, false>(ctx, Wo, out);
}

// ---------------------------------------------------------------------------
// Flash-style attention. One WG = one (b, h, 128-query tile); 8 waves x 16 rows.
// K block (contiguous 4KB in head-major layout) arrives via the Tensor Data
// Mover (wave 0 issues one 1-D D# descriptor, waits TENSORcnt, barrier
// publishes). V block is staged transposed by all threads. Online softmax.
// ---------------------------------------------------------------------------
__global__ __launch_bounds__(256) void attn_kernel(
    const unsigned short* __restrict__ Q, const unsigned short* __restrict__ Km,
    const unsigned short* __restrict__ Vm, const unsigned char* __restrict__ mask,
    const float* __restrict__ gamma, unsigned short* __restrict__ ctx) {
  __shared__ unsigned short Ks[32 * 64];       // K block, row-major [key][d]
  __shared__ unsigned short Vt[64 * 32];       // V block, transposed [d][key]
  __shared__ unsigned short Ps[8 * 16 * 32];   // per-wave P scratch (C->A relayout)

  const int tq = blockIdx.x & 15;              // S/128 = 16 query tiles
  const int bh = blockIdx.x >> 4;
  const int b = bh >> 4, h = bh & 15;          // H = 16
  const int t = threadIdx.x, lane = t & 31, w = t >> 5;
  const int lm = lane & 15, hi = lane >> 4;
  const size_t base = (size_t)bh * Sc * Dc;
  const int q0 = tq * 128 + w * 16;

  // Q A-fragments (16 rows x D=64 -> two 16x32 frags), loaded straight from global
  v16bf qf[2];
#pragma unroll
  for (int c = 0; c < 2; c++) {
    const unsigned short* qp = Q + base + (size_t)(q0 + lm) * Dc + c * 32 + hi * 16;
    qf[c] = cat8(*(const v8bf*)qp, *(const v8bf*)(qp + 8));
  }

  float mrow[8], lrow[8];
#pragma unroll
  for (int e = 0; e < 8; e++) { mrow[e] = -3.0e38f; lrow[e] = 0.0f; }
  v8f acc[4];
#pragma unroll
  for (int j = 0; j < 4; j++) acc[j] = vzero8();
  unsigned short* myP = &Ps[w * 16 * 32];

  for (int kb = 0; kb < Sc; kb += 32) {
    // ---- K block: TDM DMA, issued by wave 0 only ----
    if (w == 0) {
      const unsigned long long ga =
          (unsigned long long)(uintptr_t)(Km + base + (size_t)kb * Dc);
      const unsigned la = lds_off(&Ks[0]);
      v4u g0;
      g0[0] = 1u;                                   // count=1, user descriptor
      g0[1] = la;                                   // lds_addr
      g0[2] = (unsigned)ga;                         // global_addr[31:0]
      g0[3] = (unsigned)((ga >> 32) & 0x01FFFFFFu)  // global_addr[56:32]
              | (2u << 30);                         // type=2 ("image")
      v8i g1;
      g1[0] = 0x00010000;   // workgroup_mask=0, data_size=1 (2 bytes)
      g1[1] = 0;            // no atomic barrier; tensor_dim0[15:0]=0
      g1[2] = 0x00010002;   // tensor_dim0=131072 (hi16=2), tensor_dim1=1
      g1[3] = 0x08000000;   // tile_dim0 = 2048 elements (one 4KB line)
      g1[4] = 1;            // tile_dim1 = 1
      g1[5] = 0x00020000;   // tensor_dim0_stride = 131072
      g1[6] = 0;
      g1[7] = 0;
      v4i gz4 = {0, 0, 0, 0};
      v8i gz8 = {0, 0, 0, 0, 0, 0, 0, 0};
      __builtin_amdgcn_tensor_load_to_lds(g0, g1, gz4, gz4, gz8, 0);
      __builtin_amdgcn_s_wait_tensorcnt(0);
    }
    // ---- V block: stage transposed [d][key] ----
    {
      const int kk = t & 31, db = (t >> 5) * 8;
      const unsigned short* vp = Vm + base + (size_t)(kb + kk) * Dc + db;
      uint4 vv = *(const uint4*)vp;
      unsigned short vtmp[8];
      *(uint4*)vtmp = vv;
#pragma unroll
      for (int e = 0; e < 8; e++) Vt[(db + e) * 32 + kk] = vtmp[e];
      if (kb + 32 < Sc) __builtin_prefetch(vp + 32 * Dc, 0, 1);
    }
    __syncthreads();

    // S = Q * K^T  (16 queries x 32 keys = two 16x16 C frags, K-dim 64 = 2 steps)
    v8f s0 = vzero8(), s1 = vzero8();
#pragma unroll
    for (int c = 0; c < 2; c++) {
      v16bf b0 = *(const v16bf*)&Ks[(lm)      * 64 + c * 32 + hi * 16];
      v16bf b1 = *(const v16bf*)&Ks[(16 + lm) * 64 + c * 32 + hi * 16];
      s0 = bfwmma(qf[c], b0, s0);
      s1 = bfwmma(qf[c], b1, s1);
    }

    // scale + key-padding mask (C frag: column N == lm for both lane halves)
    const bool msk0 = mask[(size_t)b * Sc + kb + lm] != 0;
    const bool msk1 = mask[(size_t)b * Sc + kb + 16 + lm] != 0;
#pragma unroll
    for (int e = 0; e < 8; e++) {
      s0[e] = msk0 ? -3.0e38f : s0[e] * 0.125f;   // 1/sqrt(64)
      s1[e] = msk1 ? -3.0e38f : s1[e] * 0.125f;
    }

    // online softmax; row stats live per-VGPR-index (row = e + 8*hi),
    // reductions across the 16-lane half via xor shuffles 1/2/4/8
#pragma unroll
    for (int e = 0; e < 8; e++) {
      float bm = fmaxf(s0[e], s1[e]);
      bm = fmaxf(bm, __shfl_xor(bm, 1));
      bm = fmaxf(bm, __shfl_xor(bm, 2));
      bm = fmaxf(bm, __shfl_xor(bm, 4));
      bm = fmaxf(bm, __shfl_xor(bm, 8));
      const float mn = fmaxf(mrow[e], bm);
      const float sc = __expf(mrow[e] - mn);
      mrow[e] = mn;
      const float p0 = __expf(s0[e] - mn);
      const float p1 = __expf(s1[e] - mn);
      s0[e] = p0; s1[e] = p1;
      float rs = p0 + p1;
      rs += __shfl_xor(rs, 1);
      rs += __shfl_xor(rs, 2);
      rs += __shfl_xor(rs, 4);
      rs += __shfl_xor(rs, 8);
      lrow[e] = lrow[e] * sc + rs;
      acc[0][e] *= sc; acc[1][e] *= sc; acc[2][e] *= sc; acc[3][e] *= sc;
    }

    // C-layout -> A-layout relayout of P through per-wave LDS scratch
#pragma unroll
    for (int e = 0; e < 8; e++) {
      myP[(e + hi * 8) * 32 + lm]      = f2bf(s0[e]);
      myP[(e + hi * 8) * 32 + 16 + lm] = f2bf(s1[e]);
    }
    const unsigned short* pp = myP + lm * 32 + hi * 16;
    const v16bf pf = cat8(*(const v8bf*)pp, *(const v8bf*)(pp + 8));

    // O += P * V   (B frag column N = output d, contiguous K thanks to Vt)
#pragma unroll
    for (int j = 0; j < 4; j++) {
      v16bf vf = *(const v16bf*)&Vt[(j * 16 + lm) * 32 + hi * 16];
      acc[j] = bfwmma(pf, vf, acc[j]);
    }
    __syncthreads();
  }

  const float g = gamma[h];
#pragma unroll
  for (int j = 0; j < 4; j++)
#pragma unroll
    for (int e = 0; e < 8; e++) {
      const int m = q0 + e + hi * 8;
      const float val = g * acc[j][e] / lrow[e];
      ctx[((size_t)(b * Sc + m)) * Ec + h * Dc + j * 16 + lm] = f2bf(val);
    }
}

// ---------------------------------------------------------------------------
extern "C" void kernel_launch(void* const* d_in, const int* in_sizes, int n_in,
                              void* d_out, int out_size, void* d_ws, size_t ws_size,
                              hipStream_t stream) {
  const float* q = (const float*)d_in[0];
  const float* k = (const float*)d_in[1];
  const float* v = (const float*)d_in[2];
  const unsigned char* mask = (const unsigned char*)d_in[3];
  const float* Wq = (const float*)d_in[4];
  const float* Wk = (const float*)d_in[5];
  const float* Wv = (const float*)d_in[6];
  const float* Wo = (const float*)d_in[7];
  const float* gamma = (const float*)d_in[8];

  // workspace: Q,K,V head-major bf16 + ctx bf16 (4 x 16 MB = 64 MB)
  unsigned short* Qb = (unsigned short*)d_ws;
  unsigned short* Kb = Qb + (size_t)Mtot * Ec;
  unsigned short* Vb = Kb + (size_t)Mtot * Ec;
  unsigned short* Cx = Vb + (size_t)Mtot * Ec;

  dim3 blk(256);
  proj_qkv_kernel<<<dim3(Mtot / 128, Ec / 128, 3), blk, 0, stream>>>(
      q, k, v, Wq, Wk, Wv, Qb, Kb, Vb);
  attn_kernel<<<dim3(Bc * Hc * (Sc / 128)), blk, 0, stream>>>(
      Qb, Kb, Vb, mask, gamma, Cx);
  out_proj_kernel<<<dim3(Mtot / 128, Ec / 128), blk, 0, stream>>>(
      Cx, Wo, (float*)d_out);
}
